// QuadPrior_Attention_78391743086718
// MI455X (gfx1250) — matrix operand order repaired and checked by reading, not compiled, not run on hardware
//
#include <hip/hip_runtime.h>
#include <hip/hip_bf16.h>

typedef __attribute__((ext_vector_type(16))) _Float16 v16h;
typedef __attribute__((ext_vector_type(8)))  _Float16 v8h;
typedef __attribute__((ext_vector_type(8)))  float    v8f;

#define TOK   8192     // BATCH * SEQ_LEN
#define CDIM  768
#define TSEQ  2048
#define NHEAD 12
#define HD    64
#define NBLK  8
#define BS    96

union H16 { v16h v; v8h h[2]; };

__device__ __forceinline__ v8f wmma16(v16h a, v16h b, v8f c) {
  // D = A(16x32 f16) * B(32x16 f16) + C(16x16 f32)
  return __builtin_amdgcn_wmma_f32_16x16x32_f16(false, a, false, b, (short)0, c,
                                                false, false);
}

// A-tile (16x32 f16) from row-major memory; base -> (row0, k0); ld in elements.
// Lane layout: row = lane&15; halves K=[c0,c0+8) then [c0+16,c0+24), c0=8*(lane>=16).
__device__ __forceinline__ v16h load_a16(const _Float16* base, int ld, int lane) {
  const _Float16* p = base + (size_t)(lane & 15) * ld + ((lane >> 4) << 3);
  H16 u;
  u.h[0] = *(const v8h*)(p);
  u.h[1] = *(const v8h*)(p + 16);
  return u.v;
}

// B-tile (32x16 f16) from row-major memory; base -> (k0, n0); lane holds col n0+(lane&15), K as in load_a16.
__device__ __forceinline__ v16h load_b16(const _Float16* base, int ld, int lane) {
  const _Float16* p = base + (lane & 15) + (size_t)((lane >> 4) << 3) * ld;
  v16h r;
#pragma unroll
  for (int i = 0; i < 8; ++i) {
    r[i] = p[(size_t)i * ld];
    r[8 + i] = p[(size_t)(16 + i) * ld];
  }
  return r;
}

__global__ void cvt_f32_f16(const float* __restrict__ in, _Float16* __restrict__ out, int n) {
  int i = blockIdx.x * blockDim.x + threadIdx.x;
  if (i < n) out[i] = (_Float16)in[i];
}

// Pack block-tridiagonal weights into [dst_block 8][src_sel 3][96][96] f16,
// src_sel: 0 = from b-1 (Wr[b-1]), 1 = from b (Wv[b]), 2 = from b+1 (Wl[b]); edges = 0.
__global__ void combine_w(const float* __restrict__ Wv, const float* __restrict__ Wr,
                          const float* __restrict__ Wl, _Float16* __restrict__ wh) {
  int idx = blockIdx.x * blockDim.x + threadIdx.x;
  if (idx >= NBLK * 3 * BS * BS) return;
  int ij = idx % (BS * BS);
  int s  = (idx / (BS * BS)) % 3;
  int b  = idx / (3 * BS * BS);
  float v = 0.f;
  if (s == 1) v = Wv[b * BS * BS + ij];
  else if (s == 0) { if (b > 0) v = Wr[(b - 1) * BS * BS + ij]; }
  else             { if (b < NBLK - 1) v = Wl[b * BS * BS + ij]; }
  wh[idx] = (_Float16)v;
}

// One wave per (16-token tile, dst block). Output tile [16 x 96] via WMMA.
// MODE 0: f16 [tok][768]; MODE 1: f16 transposed k layout [b][h][d][t]; MODE 2: f32 [tok][768].
template <int MODE>
__global__ __launch_bounds__(32) void urdhva_proj(const _Float16* __restrict__ X,
                                                  const _Float16* __restrict__ W,
                                                  const float* __restrict__ bias,
                                                  void* __restrict__ outp) {
  const int lane = threadIdx.x;
  const int m0   = blockIdx.x * 16;   // token tile
  const int db   = blockIdx.y;        // dst block

  const v8f vzero = {0, 0, 0, 0, 0, 0, 0, 0};
  v8f acc[6];
#pragma unroll
  for (int i = 0; i < 6; ++i) acc[i] = vzero;

#pragma unroll
  for (int s = 0; s < 3; ++s) {
    const int sb = db - 1 + s;
    if (sb < 0 || sb >= NBLK) continue;
    const _Float16* wb = W + (size_t)(db * 3 + s) * BS * BS;
#pragma unroll
    for (int ks = 0; ks < 3; ++ks) {
      v16h a = load_a16(X + (size_t)m0 * CDIM + sb * BS + ks * 32, CDIM, lane);
      const _Float16* wk = wb + (size_t)ks * 32 * BS;
#pragma unroll
      for (int nt = 0; nt < 6; ++nt) {
        v16h bm = load_b16(wk + nt * 16, BS, lane);
        acc[nt] = wmma16(a, bm, acc[nt]);
      }
    }
  }

  const int n  = lane & 15;
  const int mh = (lane >> 4) << 3;
#pragma unroll
  for (int nt = 0; nt < 6; ++nt) {
    const int col = db * BS + nt * 16 + n;
    const float bb = bias[col];
#pragma unroll
    for (int r = 0; r < 8; ++r) {
      const int tok = m0 + mh + r;
      const float val = acc[nt][r] + bb;
      if (MODE == 2) {
        ((float*)outp)[(size_t)tok * CDIM + col] = val;
      } else if (MODE == 0) {
        ((_Float16*)outp)[(size_t)tok * CDIM + col] = (_Float16)val;
      } else {
        const int bidx = tok >> 11, t = tok & (TSEQ - 1);
        const int hh = col >> 6, d = col & 63;
        ((_Float16*)outp)[((size_t)((bidx * NHEAD + hh) * HD + d) << 11) + t] = (_Float16)val;
      }
    }
  }
}

// kt(idx) = qt - {0,1,2,4,8,16,32,64}[idx] == qt - ((1<<idx)>>1)
__device__ __forceinline__ int ktile_of(int qt, int idx) { return qt - ((1 << idx) >> 1); }

// One wave per (batch, head, 16-query tile). Sparse masked attention via WMMA.
__global__ __launch_bounds__(32) void sparse_attn(const _Float16* __restrict__ Q,
                                                  const _Float16* __restrict__ KT,
                                                  const _Float16* __restrict__ V,
                                                  _Float16* __restrict__ Ao) {
  __shared__ __align__(16) _Float16 P[16 * 128];
  const int lane = threadIdx.x;
  const int blk  = blockIdx.x;
  const int qt = blk & 127;
  const int hh = (blk >> 7) % NHEAD;
  const int bb = blk / (128 * NHEAD);
  const int i0 = qt * 16;

  const _Float16* qbase = Q + (size_t)(bb * TSEQ + i0) * CDIM + hh * HD;
  const v16h a0 = load_a16(qbase, CDIM, lane);
  const v16h a1 = load_a16(qbase + 32, CDIM, lane);

  const _Float16* ktb = KT + (size_t)(bb * NHEAD + hh) * HD * TSEQ;

  const v8f vzero = {0, 0, 0, 0, 0, 0, 0, 0};
  v8f sc[8];
#pragma unroll
  for (int idx = 0; idx < 8; ++idx) {
    const int kti = ktile_of(qt, idx);
    if (kti >= 0) {
      const _Float16* kb = ktb + kti * 16;                 // rows = d, cols = t
      v16h b0 = load_b16(kb, TSEQ, lane);                  // d = 0..31
      v16h b1 = load_b16(kb + (size_t)32 * TSEQ, TSEQ, lane); // d = 32..63
      v8f a = vzero;
      a = wmma16(a0, b0, a);
      a = wmma16(a1, b1, a);
      sc[idx] = a;
    } else {
      sc[idx] = vzero;
    }
  }

  const int n  = lane & 15;
  const int mh = (lane >> 4) << 3;

  // scale + analytic sparse mask: allowed iff d in [0,4] or d = power-of-two >= 8
#pragma unroll
  for (int idx = 0; idx < 8; ++idx) {
    const int kti = ktile_of(qt, idx);
    const int j = kti * 16 + n;
#pragma unroll
    for (int r = 0; r < 8; ++r) {
      const int i = i0 + mh + r;
      const int d = i - j;
      const bool ok = (kti >= 0) && (d >= 0) &&
                      (d <= 4 || (d >= 8 && (d & (d - 1)) == 0));
      sc[idx][r] = ok ? sc[idx][r] * 0.125f : -1e30f;
    }
  }

  // row softmax: row (mh+r) lives in lanes sharing lane>>4; reduce over lanes 0..15 / 16..31
#pragma unroll
  for (int r = 0; r < 8; ++r) {
    float m = -1e30f;
#pragma unroll
    for (int idx = 0; idx < 8; ++idx) m = fmaxf(m, sc[idx][r]);
#pragma unroll
    for (int w = 1; w < 16; w <<= 1) m = fmaxf(m, __shfl_xor(m, w, 32));
    float ssum = 0.f;
#pragma unroll
    for (int idx = 0; idx < 8; ++idx) {
      const float e = __expf(sc[idx][r] - m);
      sc[idx][r] = e;
      ssum += e;
    }
#pragma unroll
    for (int w = 1; w < 16; w <<= 1) ssum += __shfl_xor(ssum, w, 32);
    const float inv = 1.f / ssum;
#pragma unroll
    for (int idx = 0; idx < 8; ++idx) sc[idx][r] *= inv;
  }

  // P (C-layout f32) -> LDS f16 [16][128] for A-layout reload
#pragma unroll
  for (int idx = 0; idx < 8; ++idx)
#pragma unroll
    for (int r = 0; r < 8; ++r)
      P[(mh + r) * 128 + idx * 16 + n] = (_Float16)sc[idx][r];
  __syncthreads();

  v8f oacc[4];
#pragma unroll
  for (int ntv = 0; ntv < 4; ++ntv) oacc[ntv] = vzero;

#pragma unroll
  for (int s = 0; s < 4; ++s) {
    const _Float16* pa = &P[(size_t)(lane & 15) * 128 + s * 32 + ((lane >> 4) << 3)];
    H16 u;
    u.h[0] = *(const v8h*)(pa);
    u.h[1] = *(const v8h*)(pa + 16);
    const v16h ap = u.v;

    const int h8 = (lane >> 4) << 3;
#pragma unroll
    for (int ntv = 0; ntv < 4; ++ntv) {
      v16h w;
#pragma unroll
      for (int e = 0; e < 16; ++e) {
        const int kk = (e < 8) ? (h8 + e) : (16 + h8 + e - 8);
        const int c = s * 32 + kk;              // B row = gathered P column
        const int cidx = c >> 4;
        const int kti = ktile_of(qt, cidx);
        const int j = (kti >= 0) ? (kti * 16 + (c & 15)) : 0;  // P==0 for invalid tiles
        w[e] = V[(size_t)(bb * TSEQ + j) * CDIM + hh * HD + ntv * 16 + (lane & 15)];
      }
      oacc[ntv] = wmma16(ap, w, oacc[ntv]);
    }
  }

#pragma unroll
  for (int ntv = 0; ntv < 4; ++ntv)
#pragma unroll
    for (int r = 0; r < 8; ++r) {
      const int tok = bb * TSEQ + i0 + mh + r;
      Ao[(size_t)tok * CDIM + hh * HD + ntv * 16 + n] = (_Float16)oacc[ntv][r];
    }
}

extern "C" void kernel_launch(void* const* d_in, const int* in_sizes, int n_in,
                              void* d_out, int out_size, void* d_ws, size_t ws_size,
                              hipStream_t stream) {
  (void)in_sizes; (void)n_in; (void)out_size; (void)ws_size;
  const float* x   = (const float*)d_in[0];
  const float* qWv = (const float*)d_in[1];
  const float* qWr = (const float*)d_in[2];
  const float* qWl = (const float*)d_in[3];
  const float* qb  = (const float*)d_in[4];
  const float* kWv = (const float*)d_in[5];
  const float* kWr = (const float*)d_in[6];
  const float* kWl = (const float*)d_in[7];
  const float* kb  = (const float*)d_in[8];
  const float* vWv = (const float*)d_in[9];
  const float* vWr = (const float*)d_in[10];
  const float* vWl = (const float*)d_in[11];
  const float* vb  = (const float*)d_in[12];
  const float* oWv = (const float*)d_in[13];
  const float* oWr = (const float*)d_in[14];
  const float* oWl = (const float*)d_in[15];
  const float* ob  = (const float*)d_in[16];
  // d_in[17] = sparse_mask: reproduced analytically in-kernel.

  char* ws = (char*)d_ws;
  const size_t SZ_H = (size_t)TOK * CDIM * sizeof(_Float16);       // 12.6 MB each
  const size_t SZ_W = (size_t)NBLK * 3 * BS * BS * sizeof(_Float16);
  _Float16* xh  = (_Float16*)(ws);
  _Float16* qh  = (_Float16*)(ws + SZ_H);
  _Float16* kT  = (_Float16*)(ws + 2 * SZ_H);
  _Float16* vh  = (_Float16*)(ws + 3 * SZ_H);
  _Float16* ah  = (_Float16*)(ws + 4 * SZ_H);
  _Float16* whq = (_Float16*)(ws + 5 * SZ_H);
  _Float16* whk = (_Float16*)(ws + 5 * SZ_H + SZ_W);
  _Float16* whv = (_Float16*)(ws + 5 * SZ_H + 2 * SZ_W);
  _Float16* who = (_Float16*)(ws + 5 * SZ_H + 3 * SZ_W);

  { const int nel = TOK * CDIM;
    cvt_f32_f16<<<(nel + 255) / 256, 256, 0, stream>>>(x, xh, nel); }
  { const int nw = NBLK * 3 * BS * BS;
    const int g = (nw + 255) / 256;
    combine_w<<<g, 256, 0, stream>>>(qWv, qWr, qWl, whq);
    combine_w<<<g, 256, 0, stream>>>(kWv, kWr, kWl, whk);
    combine_w<<<g, 256, 0, stream>>>(vWv, vWr, vWl, whv);
    combine_w<<<g, 256, 0, stream>>>(oWv, oWr, oWl, who); }

  dim3 pg(TOK / 16, NBLK);
  urdhva_proj<0><<<pg, 32, 0, stream>>>(xh, whq, qb, qh);
  urdhva_proj<1><<<pg, 32, 0, stream>>>(xh, whk, kb, kT);   // k written [b][h][d][t]
  urdhva_proj<0><<<pg, 32, 0, stream>>>(xh, whv, vb, vh);

  sparse_attn<<<4 * NHEAD * (TSEQ / 16), 32, 0, stream>>>(qh, kT, vh, ah);

  urdhva_proj<2><<<pg, 32, 0, stream>>>(ah, who, ob, d_out); // final f32 output
}
